// EncoderLayer_2250562863148
// MI455X (gfx1250) — matrix operand order repaired and checked by reading, not compiled, not run on hardware
//
#include <hip/hip_runtime.h>
#include <hip/hip_bf16.h>

typedef __attribute__((ext_vector_type(16))) __bf16 v16bf;
typedef __attribute__((ext_vector_type(8)))  __bf16 v8bf;
typedef __attribute__((ext_vector_type(8)))  float  v8f;
typedef __attribute__((ext_vector_type(4)))  unsigned int v4u;
typedef __attribute__((ext_vector_type(8)))  int   v8i;
typedef __attribute__((ext_vector_type(4)))  int   v4i;

#define D_MODEL 768
#define D_FF    3072
#define NHEAD   12
#define D_K     64
#define SEQ     2048
#define BATCH   2
#define MROWS   (BATCH * SEQ)

#if __has_builtin(__builtin_amdgcn_tensor_load_to_lds)
#define USE_TDM 1
#else
#define USE_TDM 0
#endif

// ---------------------------------------------------------------------------
// Weight convert + transpose: W[K,N] f32 -> Wt[N,K] bf16 (contiguous-K rows)
// ---------------------------------------------------------------------------
__global__ __launch_bounds__(256)
void cvt_transpose_kernel(const float* __restrict__ W, __bf16* __restrict__ Wt,
                          int K, int N) {
  long long idx = (long long)blockIdx.x * 256 + threadIdx.x;
  if (idx >= (long long)K * N) return;
  int k = (int)(idx / N);
  int n = (int)(idx % N);
  Wt[(long long)n * K + k] = (__bf16)W[idx];
}

// ---------------------------------------------------------------------------
// V[B*S, 768] bf16 -> Vt[B*H, 64, S] bf16 so attention PV B-fragments are
// contiguous-K (key) loads instead of 2-byte gathers.
// ---------------------------------------------------------------------------
__global__ __launch_bounds__(256)
void transpose_v_kernel(const __bf16* __restrict__ v, __bf16* __restrict__ vt) {
  long long idx = (long long)blockIdx.x * 256 + threadIdx.x;
  if (idx >= (long long)MROWS * D_MODEL) return;
  int c = (int)(idx % D_MODEL);
  long long r = idx / D_MODEL;          // b*SEQ + s
  int b  = (int)(r / SEQ);
  int sp = (int)(r % SEQ);
  int h  = c / D_K, dk = c % D_K;
  vt[(((long long)(b * NHEAD + h) * D_K) + dk) * SEQ + sp] = v[idx];
}

// ---------------------------------------------------------------------------
// LayerNorm (torch-style: unbiased var /(n-1), eps added to STD) -> bf16
// ---------------------------------------------------------------------------
__global__ __launch_bounds__(256)
void layernorm_kernel(const float* __restrict__ x, const float* __restrict__ g,
                      const float* __restrict__ bta, __bf16* __restrict__ out) {
  const int row = blockIdx.x;
  const float* xr = x + (long long)row * D_MODEL;
  __shared__ float red[256];
  float s = 0.f;
  for (int i = threadIdx.x; i < D_MODEL; i += 256) s += xr[i];
  red[threadIdx.x] = s; __syncthreads();
  for (int d = 128; d > 0; d >>= 1) {
    if (threadIdx.x < d) red[threadIdx.x] += red[threadIdx.x + d];
    __syncthreads();
  }
  const float mean = red[0] / (float)D_MODEL;
  __syncthreads();
  float vs = 0.f;
  for (int i = threadIdx.x; i < D_MODEL; i += 256) {
    float d = xr[i] - mean; vs += d * d;
  }
  red[threadIdx.x] = vs; __syncthreads();
  for (int d = 128; d > 0; d >>= 1) {
    if (threadIdx.x < d) red[threadIdx.x] += red[threadIdx.x + d];
    __syncthreads();
  }
  const float var = red[0] / (float)(D_MODEL - 1);
  const float inv = 1.f / (sqrtf(var) + 1e-6f);
  __bf16* orow = out + (long long)row * D_MODEL;
  for (int i = threadIdx.x; i < D_MODEL; i += 256)
    orow[i] = (__bf16)(g[i] * (xr[i] - mean) * inv + bta[i]);
}

// ---------------------------------------------------------------------------
// TDM: issue async load of a 64(n) x 32(k) bf16 tile of Bt[N,K] into LDS.
// D# packed per CDNA5 ISA. TENSORcnt-tracked.
// ---------------------------------------------------------------------------
#if USE_TDM
__device__ __forceinline__
void tdm_issue_b_tile(const __bf16* gbase, long long elemOff, unsigned ldsOff,
                      int K, int N) {
  unsigned long long ga = (unsigned long long)(size_t)(gbase + elemOff);
  v4u g0;
  g0[0] = 1u;                                        // count=1, user descriptor
  g0[1] = ldsOff;                                    // lds byte address
  g0[2] = (unsigned)(ga & 0xffffffffu);              // global_addr[31:0]
  g0[3] = (unsigned)((ga >> 32) & 0x01ffffffu) | (2u << 30);  // addr[56:32], type=2
  v8i g1;
  g1[0] = (int)(1u << 16);                           // data_size=1 (2 bytes)
  g1[1] = (int)(((unsigned)K & 0xffffu) << 16);      // tensor_dim0 lo16
  g1[2] = (int)((((unsigned)K >> 16) & 0xffffu) | (((unsigned)N & 0xffffu) << 16));
  g1[3] = (int)((((unsigned)N >> 16) & 0xffffu) | (32u << 16));  // tile_dim0=32
  g1[4] = 64;                                        // tile_dim1=64, tile_dim2=0
  g1[5] = K;                                         // tensor_dim0_stride lo32
  g1[6] = 0;
  g1[7] = 0;
  v4i g2; g2[0] = 1; g2[1] = 1; g2[2] = 0; g2[3] = 0;
  v4i g3 = {0, 0, 0, 0};
#if __clang_major__ >= 23
  v8i g4 = {0, 0, 0, 0, 0, 0, 0, 0};
  __builtin_amdgcn_tensor_load_to_lds(g0, g1, g2, g3, g4, 0);
#else
  __builtin_amdgcn_tensor_load_to_lds(g0, g1, g2, g3, 0);
#endif
}
#endif

// ---------------------------------------------------------------------------
// WMMA GEMM: out[M,N] = act(oscale*(A[M,K] @ Bt[N,K]^T + bias)) (+ res)
// block = 8 waves, 128x64 block tile, wave tile 16x64 (4 accumulators).
// B tile (64x32 bf16, 4KB) staged in LDS: TDM double-buffered when available,
// else cooperative global->LDS loads. A streamed per-wave from global.
// ---------------------------------------------------------------------------
template<bool RELU, bool RESID, bool OUT_BF16>
__global__ __launch_bounds__(256)
void gemm_wmma_kernel(const __bf16* __restrict__ A,
                      const __bf16* __restrict__ Bt,
                      const float* __restrict__ bias,
                      const float* __restrict__ res,
                      float* __restrict__ outF, __bf16* __restrict__ outB,
                      int N, int K, float oscale) {
  __shared__ __bf16 ldsB[2][64 * 32];                // double-buffered B tile
  const int lane = threadIdx.x & 31;
  const int wave = threadIdx.x >> 5;
  const int half = lane >> 4;
  const int l15  = lane & 15;
  const int m0 = blockIdx.x * 128 + wave * 16;
  const int n0 = blockIdx.y * 64;

  v8f acc[4] = {};
  const __bf16* arow = A + (long long)(m0 + l15) * K;
  const int nk = K / 32;

#if USE_TDM
  const unsigned ldsOff0 = (unsigned)(size_t)(void*)(&ldsB[0][0]);
  if (wave == 0)
    tdm_issue_b_tile(Bt, (long long)n0 * K, ldsOff0, K, N);
#endif

  for (int i = 0; i < nk; ++i) {
    const int k0 = i * 32;
#if USE_TDM
    if (wave == 0) {
      if (i + 1 < nk) {
        tdm_issue_b_tile(Bt, (long long)n0 * K + (i + 1) * 32,
                         ldsOff0 + (unsigned)(((i + 1) & 1) * 64 * 32 * 2), K, N);
        __builtin_amdgcn_s_wait_tensorcnt(1);   // oldest (tile i) complete
      } else {
        __builtin_amdgcn_s_wait_tensorcnt(0);
      }
    }
    __syncthreads();
    const __bf16* bbuf = &ldsB[i & 1][0];
#else
    // fallback: 256 threads cooperatively stage the 2048-element tile
    {
      const int e = threadIdx.x * 8;               // 8 bf16 per thread
      const int nl = e >> 5, kl = e & 31;
      *(v8bf*)&ldsB[0][e] = *(const v8bf*)(Bt + (long long)(n0 + nl) * K + k0 + kl);
    }
    __syncthreads();
    const __bf16* bbuf = &ldsB[0][0];
#endif
    // A 16x32 fragment: elem e -> k = k0 + half*16 + e
    v16bf a;
    *(v8bf*)&a       = *(const v8bf*)(arow + k0 + half * 16);
    *((v8bf*)&a + 1) = *(const v8bf*)(arow + k0 + half * 16 + 8);
    __builtin_prefetch(arow + k0 + 64, 0, 0);      // global_prefetch_b8
#pragma unroll
    for (int t = 0; t < 4; ++t) {
      const __bf16* brow = bbuf + (t * 16 + l15) * 32 + half * 16;
      v16bf b;  // B 32x16 fragment: elem j -> k = k0 + half*16 + j
      *(v8bf*)&b       = *(const v8bf*)(brow);
      *((v8bf*)&b + 1) = *(const v8bf*)(brow + 8);
      acc[t] = __builtin_amdgcn_wmma_f32_16x16x32_bf16(
          false, a, false, b, (short)0, acc[t], false, false);
    }
    __syncthreads();
  }

#pragma unroll
  for (int t = 0; t < 4; ++t) {
    const int n = n0 + t * 16 + l15;
    const float bv = bias[n];
#pragma unroll
    for (int i = 0; i < 8; ++i) {
      const int m = m0 + half * 8 + i;
      float v = (acc[t][i] + bv) * oscale;
      if (RELU)  v = v > 0.f ? v : 0.f;
      if (RESID) v += res[(long long)m * N + n];
      if (OUT_BF16) outB[(long long)m * N + n] = (__bf16)v;
      else          outF[(long long)m * N + n] = v;
    }
  }
}

// ---------------------------------------------------------------------------
// Flash attention: 1 wave per 16-query tile per (b,h). Streams 32 keys/iter,
// online softmax fp32, P->A-fragment via LDS, V read from pre-transposed Vt.
// Q pre-scaled by 1/sqrt(Dk). Never materializes the SxS matrix.
// ---------------------------------------------------------------------------
__global__ __launch_bounds__(32)
void attn_kernel(const __bf16* __restrict__ Q, const __bf16* __restrict__ Kmat,
                 const __bf16* __restrict__ Vt, const int* __restrict__ mask,
                 __bf16* __restrict__ ctx) {
  __shared__ __bf16 lds[16 * 32];
  const int lane = threadIdx.x;
  const int half = lane >> 4;
  const int l15  = lane & 15;
  const int q0 = blockIdx.x * 16;
  const int bh = blockIdx.y;
  const int b  = bh / NHEAD;
  const int hh = bh % NHEAD;
  const long long rs = D_MODEL;
  const __bf16* Qb  = Q    + (long long)b * SEQ * rs + hh * D_K;
  const __bf16* Kb  = Kmat + (long long)b * SEQ * rs + hh * D_K;
  const __bf16* Vtb = Vt   + (long long)bh * D_K * SEQ;   // [64, SEQ]
  const int* maskb = mask + b * SEQ;

  v16bf aq[2];
  {
    const __bf16* qrow = Qb + (long long)(q0 + l15) * rs;
#pragma unroll
    for (int c = 0; c < 2; ++c) {
      *(v8bf*)&aq[c]       = *(const v8bf*)(qrow + c * 32 + half * 16);
      *((v8bf*)&aq[c] + 1) = *(const v8bf*)(qrow + c * 32 + half * 16 + 8);
    }
  }

  float mrow[8], lrow[8];
#pragma unroll
  for (int i = 0; i < 8; ++i) { mrow[i] = -3.0e38f; lrow[i] = 0.f; }
  v8f o[4] = {};

  for (int kb = 0; kb < SEQ; kb += 32) {
    v8f s[2];
#pragma unroll
    for (int t = 0; t < 2; ++t) {
      const int key = kb + t * 16 + l15;
      const __bf16* krow = Kb + (long long)key * rs;
      v8f cacc = {};
#pragma unroll
      for (int c = 0; c < 2; ++c) {
        v16bf bk;
        *(v8bf*)&bk       = *(const v8bf*)(krow + c * 32 + half * 16);
        *((v8bf*)&bk + 1) = *(const v8bf*)(krow + c * 32 + half * 16 + 8);
        cacc = __builtin_amdgcn_wmma_f32_16x16x32_bf16(
            false, aq[c], false, bk, (short)0, cacc, false, false);
      }
      const float madd = (maskb[key] == 0) ? -1.0e9f : 0.f;
#pragma unroll
      for (int i = 0; i < 8; ++i) cacc[i] += madd;
      s[t] = cacc;
    }
    float rm[8];
#pragma unroll
    for (int i = 0; i < 8; ++i) rm[i] = fmaxf(s[0][i], s[1][i]);
#pragma unroll
    for (int d = 1; d < 16; d <<= 1)
#pragma unroll
      for (int i = 0; i < 8; ++i) rm[i] = fmaxf(rm[i], __shfl_xor(rm[i], d, 32));
    float scl[8], ps[8];
#pragma unroll
    for (int i = 0; i < 8; ++i) {
      const float mn = fmaxf(mrow[i], rm[i]);
      scl[i] = __expf(mrow[i] - mn);
      mrow[i] = mn;
      const float p0 = __expf(s[0][i] - mn);
      const float p1 = __expf(s[1][i] - mn);
      s[0][i] = p0; s[1][i] = p1;
      ps[i] = p0 + p1;
    }
#pragma unroll
    for (int d = 1; d < 16; d <<= 1)
#pragma unroll
      for (int i = 0; i < 8; ++i) ps[i] += __shfl_xor(ps[i], d, 32);
#pragma unroll
    for (int i = 0; i < 8; ++i) lrow[i] = lrow[i] * scl[i] + ps[i];
#pragma unroll
    for (int t = 0; t < 4; ++t)
#pragma unroll
      for (int i = 0; i < 8; ++i) o[t][i] *= scl[i];

    // P (16x32, C-layout) -> LDS -> A-fragment layout
    __syncthreads();
#pragma unroll
    for (int t = 0; t < 2; ++t)
#pragma unroll
      for (int i = 0; i < 8; ++i)
        lds[(i + 8 * half) * 32 + t * 16 + l15] = (__bf16)s[t][i];
    __syncthreads();
    v16bf pa;
    {
      const __bf16* prow = lds + l15 * 32;
      *(v8bf*)&pa       = *(const v8bf*)(prow + half * 16);
      *((v8bf*)&pa + 1) = *(const v8bf*)(prow + half * 16 + 8);
    }
    // o += P @ V : V B-fragment rows contiguous from Vt[dk, key]
#pragma unroll
    for (int t = 0; t < 4; ++t) {
      const __bf16* vrow = Vtb + (long long)(t * 16 + l15) * SEQ + kb + half * 16;
      v16bf bv;
      *(v8bf*)&bv       = *(const v8bf*)(vrow);
      *((v8bf*)&bv + 1) = *(const v8bf*)(vrow + 8);
      o[t] = __builtin_amdgcn_wmma_f32_16x16x32_bf16(
          false, pa, false, bv, (short)0, o[t], false, false);
    }
  }

#pragma unroll
  for (int t = 0; t < 4; ++t)
#pragma unroll
    for (int i = 0; i < 8; ++i) {
      const float v = o[t][i] / lrow[i];
      ctx[((long long)b * SEQ + q0 + i + 8 * half) * rs + hh * D_K + t * 16 + l15] =
          (__bf16)v;
    }
}

// ---------------------------------------------------------------------------
extern "C" void kernel_launch(void* const* d_in, const int* in_sizes, int n_in,
                              void* d_out, int out_size, void* d_ws, size_t ws_size,
                              hipStream_t stream) {
  (void)in_sizes; (void)n_in; (void)out_size; (void)ws_size;
  const float* x    = (const float*)d_in[0];
  const int*   mask = (const int*)  d_in[1];
  const float* Wq = (const float*)d_in[2];  const float* bq = (const float*)d_in[3];
  const float* Wk = (const float*)d_in[4];  const float* bk = (const float*)d_in[5];
  const float* Wv = (const float*)d_in[6];  const float* bv = (const float*)d_in[7];
  const float* Wo = (const float*)d_in[8];  const float* bo = (const float*)d_in[9];
  const float* ln1a = (const float*)d_in[10]; const float* ln1b = (const float*)d_in[11];
  const float* W1 = (const float*)d_in[12]; const float* b1 = (const float*)d_in[13];
  const float* W2 = (const float*)d_in[14]; const float* b2 = (const float*)d_in[15];
  const float* ln2a = (const float*)d_in[16]; const float* ln2b = (const float*)d_in[17];

  char* ws = (char*)d_ws;
  size_t off = 0;
  auto alloc = [&](size_t bytes) -> void* {
    void* p = ws + off;
    off += (bytes + 255) & ~(size_t)255;
    return p;
  };
  __bf16* wqT = (__bf16*)alloc((size_t)D_MODEL * D_MODEL * 2);
  __bf16* wkT = (__bf16*)alloc((size_t)D_MODEL * D_MODEL * 2);
  __bf16* wvT = (__bf16*)alloc((size_t)D_MODEL * D_MODEL * 2);
  __bf16* woT = (__bf16*)alloc((size_t)D_MODEL * D_MODEL * 2);
  __bf16* w1T = (__bf16*)alloc((size_t)D_MODEL * D_FF * 2);
  __bf16* w2T = (__bf16*)alloc((size_t)D_FF * D_MODEL * 2);
  __bf16* h   = (__bf16*)alloc((size_t)MROWS * D_MODEL * 2);   // h1, reused as h2
  __bf16* q   = (__bf16*)alloc((size_t)MROWS * D_MODEL * 2);
  __bf16* k   = (__bf16*)alloc((size_t)MROWS * D_MODEL * 2);
  __bf16* v   = (__bf16*)alloc((size_t)MROWS * D_MODEL * 2);
  __bf16* ctx = (__bf16*)alloc((size_t)MROWS * D_MODEL * 2);
  __bf16* vt  = (__bf16*)alloc((size_t)MROWS * D_MODEL * 2);   // [B*H, 64, SEQ]
  float*  y   = (float*) alloc((size_t)MROWS * D_MODEL * 4);
  __bf16* ff  = q;  // q..ctx region (4 * M*768*2 == M*3072*2) reused after O-proj

  const int ncvt = 256;
  const long long eDD = (long long)D_MODEL * D_MODEL;
  const long long eDF = (long long)D_MODEL * D_FF;
  const long long eMD = (long long)MROWS * D_MODEL;
  // 1) weights -> bf16, transposed [N,K]
  cvt_transpose_kernel<<<(unsigned)((eDD + ncvt - 1) / ncvt), ncvt, 0, stream>>>(Wq, wqT, D_MODEL, D_MODEL);
  cvt_transpose_kernel<<<(unsigned)((eDD + ncvt - 1) / ncvt), ncvt, 0, stream>>>(Wk, wkT, D_MODEL, D_MODEL);
  cvt_transpose_kernel<<<(unsigned)((eDD + ncvt - 1) / ncvt), ncvt, 0, stream>>>(Wv, wvT, D_MODEL, D_MODEL);
  cvt_transpose_kernel<<<(unsigned)((eDD + ncvt - 1) / ncvt), ncvt, 0, stream>>>(Wo, woT, D_MODEL, D_MODEL);
  cvt_transpose_kernel<<<(unsigned)((eDF + ncvt - 1) / ncvt), ncvt, 0, stream>>>(W1, w1T, D_MODEL, D_FF);
  cvt_transpose_kernel<<<(unsigned)((eDF + ncvt - 1) / ncvt), ncvt, 0, stream>>>(W2, w2T, D_FF, D_MODEL);

  // 2) LN1
  layernorm_kernel<<<MROWS, 256, 0, stream>>>(x, ln1a, ln1b, h);

  // 3) QKV projections (q folded with 1/sqrt(Dk))
  dim3 gblk(256);
  dim3 ggrid(MROWS / 128, D_MODEL / 64);
  gemm_wmma_kernel<false, false, true><<<ggrid, gblk, 0, stream>>>(h, wqT, bq, nullptr, nullptr, q, D_MODEL, D_MODEL, 0.125f);
  gemm_wmma_kernel<false, false, true><<<ggrid, gblk, 0, stream>>>(h, wkT, bk, nullptr, nullptr, k, D_MODEL, D_MODEL, 1.0f);
  gemm_wmma_kernel<false, false, true><<<ggrid, gblk, 0, stream>>>(h, wvT, bv, nullptr, nullptr, v, D_MODEL, D_MODEL, 1.0f);

  // 3b) V -> Vt [B*H, Dk, SEQ]
  transpose_v_kernel<<<(unsigned)((eMD + ncvt - 1) / ncvt), ncvt, 0, stream>>>(v, vt);

  // 4) flash attention
  dim3 agrid(SEQ / 16, BATCH * NHEAD);
  attn_kernel<<<agrid, 32, 0, stream>>>(q, k, vt, mask, ctx);

  // 5) output projection + residual(x) -> y (fp32)
  gemm_wmma_kernel<false, true, false><<<ggrid, gblk, 0, stream>>>(ctx, woT, bo, x, y, nullptr, D_MODEL, D_MODEL, 1.0f);

  // 6) LN2 -> h2 (reuse h)
  layernorm_kernel<<<MROWS, 256, 0, stream>>>(y, ln2a, ln2b, h);

  // 7) FFN1: relu(h2 @ W1 + b1) -> ff bf16 [M, 3072]
  dim3 fgrid1(MROWS / 128, D_FF / 64);
  gemm_wmma_kernel<true, false, true><<<fgrid1, gblk, 0, stream>>>(h, w1T, b1, nullptr, nullptr, ff, D_FF, D_MODEL, 1.0f);

  // 8) FFN2: y + ff @ W2 + b2 -> d_out fp32
  gemm_wmma_kernel<false, true, false><<<ggrid, gblk, 0, stream>>>(ff, w2T, b2, y, (float*)d_out, nullptr, D_MODEL, D_FF, 1.0f);
}
